// AdaptiveGraphLearner_59502476919383
// MI455X (gfx1250) — matrix operand order repaired and checked by reading, not compiled, not run on hardware
//
#include <hip/hip_runtime.h>
#include <hip/hip_bf16.h>

// ---------------------------------------------------------------------------
// AdaptiveGraphLearner for MI455X (gfx1250, wave32, WMMA).
// All GEMMs use v_wmma_f32_16x16x32_bf16 (bf16 in, f32 accumulate).
// ---------------------------------------------------------------------------

typedef __attribute__((ext_vector_type(16))) __bf16 v16bf;
typedef __attribute__((ext_vector_type(8)))  float  v8f;

constexpr int NN     = 512;
constexpr int BB     = 8;
constexpr int HH     = 64;
constexpr int K_TOP  = 26163;              // int(0.1 * 512 * 511)
constexpr int NSQ    = NN * NN;            // 262144

// ---- WMMA bf16 fragment index helpers (CDNA5 ISA 7.12.2 layouts) ----------
// A (16x32, MxK): lanes 0-15 -> M=lane, K pairs {0..7,16..23}; lanes 16-31 ->
// M=lane-16, K pairs {8..15,24..31}.
__device__ __forceinline__ int a_k_of(int lane, int v) {
    int kb = (lane & 16) ? 8 : 0;
    return (v < 4) ? (kb + 2 * v) : (16 + kb + 2 * (v - 4));
}
// B (32x16, KxN): lanes 0-15 -> N=lane, K {0..7,16..23}; lanes 16-31 -> N=lane-16, K {8..15,24..31}.
__device__ __forceinline__ int b_k_of(int lane, int v) {
    int kb = (lane & 16) ? 8 : 0;
    return (v < 4) ? (kb + 2 * v) : (16 + kb + 2 * (v - 4));
}
// C/D (16x16): lane holds N = lane&15; VGPR r holds M = r + (lane>=16 ? 8 : 0).

// ---------------------------------------------------------------------------
// Kernel 1: emb = relu(x*w_emb + b_emb) @ w_proj + b_proj   (4096x64, bf16 out)
// One wave per 16-row M tile; K=64 -> 2 WMMA per 16-col N tile, 4 N tiles.
// ---------------------------------------------------------------------------
__global__ __launch_bounds__(128) void k_emb(const float* __restrict__ x,
                                             const float* __restrict__ w_emb,
                                             const float* __restrict__ b_emb,
                                             const float* __restrict__ w_proj,
                                             const float* __restrict__ b_proj,
                                             __bf16* __restrict__ emb_out) {
    const int wave = threadIdx.x >> 5;
    const int lane = threadIdx.x & 31;
    const int row0 = (blockIdx.x * 4 + wave) * 16;       // 256 tiles total
    const int r    = row0 + (lane & 15);
    const float xr = x[r];

    v16bf a_lo, a_hi;
#pragma unroll
    for (int v = 0; v < 8; ++v) {
        const int k = a_k_of(lane, v);
        a_lo[2 * v]     = (__bf16)fmaxf(xr * w_emb[k]      + b_emb[k],      0.f);
        a_lo[2 * v + 1] = (__bf16)fmaxf(xr * w_emb[k + 1]  + b_emb[k + 1],  0.f);
        a_hi[2 * v]     = (__bf16)fmaxf(xr * w_emb[k + 32] + b_emb[k + 32], 0.f);
        a_hi[2 * v + 1] = (__bf16)fmaxf(xr * w_emb[k + 33] + b_emb[k + 33], 0.f);
    }

#pragma unroll
    for (int n0 = 0; n0 < HH; n0 += 16) {
        const int n = n0 + (lane & 15);
        v8f c = {};
#pragma unroll
        for (int ks = 0; ks < 2; ++ks) {
            v16bf bm;
#pragma unroll
            for (int v = 0; v < 8; ++v) {
                const int k = ks * 32 + b_k_of(lane, v);
                bm[2 * v]     = (__bf16)w_proj[k * HH + n];
                bm[2 * v + 1] = (__bf16)w_proj[(k + 1) * HH + n];
            }
            c = __builtin_amdgcn_wmma_f32_16x16x32_bf16(
                    false, (ks == 0 ? a_lo : a_hi), false, bm,
                    (short)0, c, false, false);
        }
        const float bp = b_proj[n];
#pragma unroll
        for (int rr = 0; rr < 8; ++rr) {
            const int m = rr + ((lane & 16) ? 8 : 0);
            emb_out[(row0 + m) * HH + n] = (__bf16)(c[rr] + bp);
        }
    }
}

// ---------------------------------------------------------------------------
// Kernel 2: ai = emb @ w1[:64], aj = emb @ w1[64:]   (fp32 out, no bias yet)
// ---------------------------------------------------------------------------
__global__ __launch_bounds__(128) void k_aiaj(const __bf16* __restrict__ emb,
                                              const float* __restrict__ w1,
                                              float* __restrict__ ai,
                                              float* __restrict__ aj) {
    const int wave = threadIdx.x >> 5;
    const int lane = threadIdx.x & 31;
    const int row0 = (blockIdx.x * 4 + wave) * 16;
    const int r    = row0 + (lane & 15);

    v16bf a_lo, a_hi;
#pragma unroll
    for (int v = 0; v < 8; ++v) {
        const int k = a_k_of(lane, v);
        a_lo[2 * v]     = emb[r * HH + k];
        a_lo[2 * v + 1] = emb[r * HH + k + 1];
        a_hi[2 * v]     = emb[r * HH + k + 32];
        a_hi[2 * v + 1] = emb[r * HH + k + 33];
    }

#pragma unroll
    for (int half = 0; half < 2; ++half) {
        const int koff = half * HH;                       // w1a rows 0..63, w1b rows 64..127
        float* __restrict__ out = half ? aj : ai;
#pragma unroll
        for (int n0 = 0; n0 < HH; n0 += 16) {
            const int n = n0 + (lane & 15);
            v8f c = {};
#pragma unroll
            for (int ks = 0; ks < 2; ++ks) {
                v16bf bm;
#pragma unroll
                for (int v = 0; v < 8; ++v) {
                    const int k = koff + ks * 32 + b_k_of(lane, v);
                    bm[2 * v]     = (__bf16)w1[k * HH + n];
                    bm[2 * v + 1] = (__bf16)w1[(k + 1) * HH + n];
                }
                c = __builtin_amdgcn_wmma_f32_16x16x32_bf16(
                        false, (ks == 0 ? a_lo : a_hi), false, bm,
                        (short)0, c, false, false);
            }
#pragma unroll
            for (int rr = 0; rr < 8; ++rr) {
                const int m = rr + ((lane & 16) ? 8 : 0);
                out[(row0 + m) * HH + n] = c[rr];
            }
        }
    }
}

// ---------------------------------------------------------------------------
// Kernel 3: fused pairwise MLP. One block = one 16x16 (i,j) tile of one batch.
// For M-tile t (fixed i = t, j = lane&15): A[m][k] = relu(ai[i][k]+aj[m][k]+b1[k]).
// h2 via 4 WMMAs (2 N-tiles x K=64), then 32->1 dot + sigmoid in registers.
// ---------------------------------------------------------------------------
__global__ __launch_bounds__(256) void k_pair(const float* __restrict__ ai,
                                              const float* __restrict__ aj,
                                              const float* __restrict__ b1,
                                              const float* __restrict__ w2,
                                              const float* __restrict__ b2,
                                              const float* __restrict__ w3,
                                              const float* __restrict__ b3,
                                              float* __restrict__ probs_raw) {
    __shared__ float s_ai[16 * HH];
    __shared__ float s_aj[16 * HH];
    __shared__ float s_b1[HH];
    __shared__ float s_b2[32];
    __shared__ float s_w3[32];
    __shared__ float s_b3;

    const int blk = blockIdx.x;
    const int b   = blk >> 10;           // 32*32 tiles per batch
    const int it  = (blk >> 5) & 31;
    const int jt  = blk & 31;
    const int i0  = it * 16, j0 = jt * 16;
    const int tid = threadIdx.x;

    for (int e = tid; e < 16 * HH; e += 256) {
        const int ti = e >> 6, k = e & 63;
        s_ai[e] = ai[(b * NN + i0 + ti) * HH + k];
        s_aj[e] = aj[(b * NN + j0 + ti) * HH + k];
    }
    if (tid < 64)                 s_b1[tid]      = b1[tid];
    else if (tid < 96)            s_b2[tid - 64] = b2[tid - 64];
    else if (tid < 128)           s_w3[tid - 96] = w3[tid - 96];
    else if (tid == 128)          s_b3           = b3[0];
    __syncthreads();

    const int wave = tid >> 5;
    const int lane = tid & 31;
    const int nl   = lane & 15;

    // w2 fragments: [N half][K step], built once per wave.
    v16bf bw2[2][2];
#pragma unroll
    for (int nh = 0; nh < 2; ++nh)
#pragma unroll
        for (int ks = 0; ks < 2; ++ks) {
            const int n = nh * 16 + nl;
#pragma unroll
            for (int v = 0; v < 8; ++v) {
                const int k = ks * 32 + b_k_of(lane, v);
                bw2[nh][ks][2 * v]     = (__bf16)w2[k * 32 + n];
                bw2[nh][ks][2 * v + 1] = (__bf16)w2[(k + 1) * 32 + n];
            }
        }
    const float w3lo = s_w3[nl],      w3hi = s_w3[16 + nl];
    const float b2lo = s_b2[nl],      b2hi = s_b2[16 + nl];

#pragma unroll
    for (int tt = 0; tt < 2; ++tt) {
        const int t = wave * 2 + tt;                      // i_local
        v16bf a_lo, a_hi;
#pragma unroll
        for (int v = 0; v < 8; ++v) {
            const int k = a_k_of(lane, v);
            a_lo[2*v]   = (__bf16)fmaxf(s_ai[t*HH + k]    + s_aj[nl*HH + k]    + s_b1[k],    0.f);
            a_lo[2*v+1] = (__bf16)fmaxf(s_ai[t*HH + k+1]  + s_aj[nl*HH + k+1]  + s_b1[k+1],  0.f);
            a_hi[2*v]   = (__bf16)fmaxf(s_ai[t*HH + k+32] + s_aj[nl*HH + k+32] + s_b1[k+32], 0.f);
            a_hi[2*v+1] = (__bf16)fmaxf(s_ai[t*HH + k+33] + s_aj[nl*HH + k+33] + s_b1[k+33], 0.f);
        }
        v8f c0 = {}, c1 = {};
        c0 = __builtin_amdgcn_wmma_f32_16x16x32_bf16(false, a_lo, false, bw2[0][0], (short)0, c0, false, false);
        c0 = __builtin_amdgcn_wmma_f32_16x16x32_bf16(false, a_hi, false, bw2[0][1], (short)0, c0, false, false);
        c1 = __builtin_amdgcn_wmma_f32_16x16x32_bf16(false, a_lo, false, bw2[1][0], (short)0, c1, false, false);
        c1 = __builtin_amdgcn_wmma_f32_16x16x32_bf16(false, a_hi, false, bw2[1][1], (short)0, c1, false, false);

        // dot with w3 over n (this lane owns n=nl and n=nl+16 for 8 rows)
        float dot[8];
#pragma unroll
        for (int rr = 0; rr < 8; ++rr)
            dot[rr] = fmaxf(c0[rr] + b2lo, 0.f) * w3lo +
                      fmaxf(c1[rr] + b2hi, 0.f) * w3hi;
        // reduce over the 16 lanes of each half (xor masks stay in-half)
#pragma unroll
        for (int m = 1; m < 16; m <<= 1)
#pragma unroll
            for (int rr = 0; rr < 8; ++rr)
                dot[rr] += __shfl_xor(dot[rr], m, 32);

        if (nl < 8) {
            float val = dot[0];
#pragma unroll
            for (int rr = 1; rr < 8; ++rr) val = (nl == rr) ? dot[rr] : val;
            const int mrow = (lane & 7) + ((lane & 16) ? 8 : 0);   // j_local
            const float p  = 1.f / (1.f + __expf(-(val + s_b3)));
            probs_raw[((size_t)b * NN + i0 + t) * NN + j0 + mrow] = p;
        }
    }
}

// ---------------------------------------------------------------------------
// Kernel 4: symmetrize + zero diag; write into d_out's adj region (temp) and
// build per-batch 1024-bin histogram (integer atomics -> deterministic).
// 512 blocks: one batch-row-group (8 rows) each.
// ---------------------------------------------------------------------------
__global__ __launch_bounds__(256) void k_sym_hist(const float* __restrict__ raw,
                                                  float* __restrict__ sym,
                                                  int* __restrict__ hist) {
    __shared__ int lh[1024];
    const int blk = blockIdx.x;
    const int b   = blk >> 6;
    const int r0  = (blk & 63) * 8;
    for (int e = threadIdx.x; e < 1024; e += 256) lh[e] = 0;
    __syncthreads();
    const float* base = raw + (size_t)b * NSQ;
    for (int e = threadIdx.x; e < 8 * NN; e += 256) {
        const int i = r0 + (e >> 9);
        const int j = e & (NN - 1);
        const float v = (i == j) ? 0.f : 0.5f * (base[i * NN + j] + base[j * NN + i]);
        sym[(size_t)b * NSQ + i * NN + j] = v;
        int bin = (int)(v * 1024.f);
        bin = bin < 0 ? 0 : (bin > 1023 ? 1023 : bin);
        atomicAdd(&lh[bin], 1);
    }
    __syncthreads();
    for (int e = threadIdx.x; e < 1024; e += 256)
        if (lh[e]) atomicAdd(&hist[b * 1024 + e], lh[e]);
}

// ---------------------------------------------------------------------------
// Kernel 5: find threshold bin + tie quota per batch.
// ---------------------------------------------------------------------------
__global__ void k_select(const int* __restrict__ hist, int* __restrict__ sel) {
    const int b = blockIdx.x;
    if (threadIdx.x == 0) {
        int cum = 0, tbin = 0, above = 0;
        for (int bin = 1023; bin >= 0; --bin) {
            const int h = hist[b * 1024 + bin];
            if (cum + h >= K_TOP) { tbin = bin; above = cum; break; }
            cum += h;
        }
        sel[b * 2]     = tbin;
        sel[b * 2 + 1] = K_TOP - above;   // how many to take from the tie bin
    }
}

// ---------------------------------------------------------------------------
// Kernel 6: deterministic ordered compaction. One 1024-thread block per batch
// scans 256 chunks sequentially, emitting exactly K_TOP ascending indices:
// all elements strictly above the threshold bin, plus the first `quota`
// elements (in index order) inside it. Rewrites adj in place (else 0).
// ---------------------------------------------------------------------------
__global__ __launch_bounds__(1024) void k_compact(const float* __restrict__ sym,
                                                  const int* __restrict__ sel,
                                                  float* __restrict__ out_rows,
                                                  float* __restrict__ out_cols,
                                                  float* __restrict__ out_w,
                                                  float* __restrict__ adj) {
    __shared__ int wTotB[32], wTotT[32], wPreB[32], wPreT[32];
    __shared__ int runB, runOut, chunkB, chunkT;
    const int b    = blockIdx.x;
    const int tid  = threadIdx.x;
    const int wave = tid >> 5;
    const int lane = tid & 31;
    const int tbin  = sel[b * 2];
    const int quota = sel[b * 2 + 1];
    if (tid == 0) { runB = 0; runOut = 0; }
    __syncthreads();
    const float* base = sym + (size_t)b * NSQ;

    for (int chunk = 0; chunk < NSQ / 1024; ++chunk) {
        const int e = chunk * 1024 + tid;
        const float v = base[e];
        int bin = (int)(v * 1024.f);
        bin = bin < 0 ? 0 : (bin > 1023 ? 1023 : bin);
        const bool isB = (bin == tbin);
        const bool isA = (bin > tbin);

        const unsigned int mb = (unsigned int)__ballot(isB);
        const int rB = __popc(mb & ((1u << lane) - 1u));
        if (lane == 0) wTotB[wave] = __popc(mb);
        __syncthreads();
        if (tid == 0) {
            int s = 0;
            for (int w = 0; w < 32; ++w) { wPreB[w] = s; s += wTotB[w]; }
            chunkB = s;
        }
        __syncthreads();

        const bool take = isA || (isB && (runB + wPreB[wave] + rB) < quota);
        const unsigned int mt = (unsigned int)__ballot(take);
        const int rT = __popc(mt & ((1u << lane) - 1u));
        if (lane == 0) wTotT[wave] = __popc(mt);
        __syncthreads();
        if (tid == 0) {
            int s = 0;
            for (int w = 0; w < 32; ++w) { wPreT[w] = s; s += wTotT[w]; }
            chunkT = s;
        }
        __syncthreads();

        if (take) {
            const int pos = runOut + wPreT[wave] + rT;
            if (pos < K_TOP) {
                const size_t g = (size_t)b * K_TOP + pos;
                out_rows[g] = (float)(e / NN + b * NN);
                out_cols[g] = (float)((e & (NN - 1)) + b * NN);
                out_w[g]    = v;
            }
        }
        adj[(size_t)b * NSQ + e] = take ? v : 0.f;   // in-place rewrite of sym

        __syncthreads();
        if (tid == 0) { runB += chunkB; runOut += chunkT; }
        __syncthreads();
    }
}

// ---------------------------------------------------------------------------
__global__ void k_zero_hist(int* __restrict__ hist, int* __restrict__ sel) {
    const int i = blockIdx.x * blockDim.x + threadIdx.x;
    if (i < BB * 1024) hist[i] = 0;
    if (i < BB * 2)    sel[i]  = 0;
}

// ---------------------------------------------------------------------------
extern "C" void kernel_launch(void* const* d_in, const int* in_sizes, int n_in,
                              void* d_out, int out_size, void* d_ws, size_t ws_size,
                              hipStream_t stream) {
    const float* x      = (const float*)d_in[0];
    const float* w_emb  = (const float*)d_in[1];
    const float* b_emb  = (const float*)d_in[2];
    const float* w_proj = (const float*)d_in[3];
    const float* b_proj = (const float*)d_in[4];
    const float* w1     = (const float*)d_in[5];
    const float* b1     = (const float*)d_in[6];
    const float* w2     = (const float*)d_in[7];
    const float* b2     = (const float*)d_in[8];
    const float* w3     = (const float*)d_in[9];
    const float* b3     = (const float*)d_in[10];
    // d_in[11] = batch indices: implied by layout, unused.

    // Workspace layout (~10.6 MB total)
    char* ws = (char*)d_ws;
    __bf16* ws_emb = (__bf16*)(ws);                         // 4096*64*2  = 512 KB
    float*  ws_ai  = (float*)(ws + (512u << 10));           // 1 MB
    float*  ws_aj  = (float*)(ws + (512u << 10) + (1u << 20));
    float*  ws_raw = (float*)(ws + (512u << 10) + (2u << 20));   // 8 MB
    int*    ws_hist = (int*)(ws + (512u << 10) + (2u << 20) + (size_t)BB * NSQ * 4);
    int*    ws_sel  = (int*)((char*)ws_hist + BB * 1024 * 4);

    // Output layout: rows | cols | weights | adj
    const size_t BK = (size_t)BB * K_TOP;
    float* out_rows = (float*)d_out;
    float* out_cols = out_rows + BK;
    float* out_w    = out_rows + 2 * BK;
    float* out_adj  = out_rows + 3 * BK;

    k_zero_hist<<<(BB * 1024 + 255) / 256, 256, 0, stream>>>(ws_hist, ws_sel);
    k_emb <<<64, 128, 0, stream>>>(x, w_emb, b_emb, w_proj, b_proj, ws_emb);
    k_aiaj<<<64, 128, 0, stream>>>(ws_emb, w1, ws_ai, ws_aj);
    k_pair<<<BB * 32 * 32, 256, 0, stream>>>(ws_ai, ws_aj, b1, w2, b2, w3, b3, ws_raw);
    k_sym_hist<<<BB * 64, 256, 0, stream>>>(ws_raw, out_adj, ws_hist);
    k_select<<<BB, 32, 0, stream>>>(ws_hist, ws_sel);
    k_compact<<<BB, 1024, 0, stream>>>(out_adj, ws_sel, out_rows, out_cols, out_w, out_adj);
}
